// Gpt2Attention_16372415332940
// MI455X (gfx1250) — matrix operand order repaired and checked by reading, not compiled, not run on hardware
//
#include <hip/hip_runtime.h>
#include <hip/hip_bf16.h>

// GPT-2 attention block for gfx1250 (MI455X), bf16 WMMA with f32 accumulate.
// Pipeline: qkv_kernel (GEMM+split/scale/transpose-V) -> attn_kernel (flash
// attention, transposed-score trick) -> proj_kernel (GEMM+bias, f32 out).
// All tile/fragment traffic is 16-byte vectorized (global b128, ds b128);
// f32->bf16 uses the packed hardware convert when the toolchain exposes it.

typedef __attribute__((ext_vector_type(16))) __bf16 v16bf;
typedef __attribute__((ext_vector_type(8)))  float  v8f;
typedef __attribute__((ext_vector_type(4)))  unsigned uint4v;

union FragB16 { v16bf v; unsigned u[8]; uint4v q[2]; unsigned short h[16]; };
union FragF32 { v8f  v; float f[8]; };

#if defined(__has_builtin)
#if __has_builtin(__builtin_amdgcn_cvt_pk_bf16_f32)
#define HAVE_CVT_PK_BF16 1
#endif
#endif

__device__ __forceinline__ unsigned packbf(float lo, float hi) {
#ifdef HAVE_CVT_PK_BF16
  auto r = __builtin_amdgcn_cvt_pk_bf16_f32(lo, hi);   // v_cvt_pk_bf16_f32
  return __builtin_bit_cast(unsigned, r);
#else
  unsigned a = __builtin_bit_cast(unsigned, lo);
  unsigned b = __builtin_bit_cast(unsigned, hi);
  a += 0x7FFFu + ((a >> 16) & 1u);
  b += 0x7FFFu + ((b >> 16) & 1u);
  return (a >> 16) | (b & 0xFFFF0000u);
#endif
}
__device__ __forceinline__ unsigned short f2bf(float x) {
  return (unsigned short)(packbf(x, 0.f) & 0xFFFFu);
}
__device__ __forceinline__ v8f wmma_bf16(const FragB16& a, const FragB16& b, v8f c) {
  // D = A(16x32 bf16) * B(32x16 bf16) + C(16x16 f32)
  return __builtin_amdgcn_wmma_f32_16x16x32_bf16(false, a.v, false, b.v,
                                                 (short)0, c, false, false);
}
// Load a 16-VGPR-half fragment as two b128 vectors. p0/p1 are 16B-aligned.
__device__ __forceinline__ void load_frag(FragB16& f, const unsigned short* p0,
                                          const unsigned short* p1) {
  f.q[0] = *(const uint4v*)p0;
  f.q[1] = *(const uint4v*)p1;
}

// ---------------------------------------------------------------------------
#define BB 2
#define SS 2048
#define DD 2048
#define HH 16
#define HD 128
#define M1 (BB*SS)        // 4096
#define N1 (3*HH*HD)      // 6144

// ---------------------------------------------------------------------------
// Kernel 1: QKV projection. X[4096][2048] * W[2048][6144] + bias.
// Block computes 128x64 tile; 8 waves, each a 32x32 sub-tile (2x2 WMMA).
// Epilogue routes to Q (scaled, [b][h][s][d]), K ([b][h][s][d]),
// Vt (transposed, [b][h][d][s]) — all bf16.
// ---------------------------------------------------------------------------
__global__ __launch_bounds__(256) void qkv_kernel(
    const float* __restrict__ X, const float* __restrict__ W,
    const float* __restrict__ bias, const int* __restrict__ layer_idx,
    unsigned short* __restrict__ Qg, unsigned short* __restrict__ Kg,
    unsigned short* __restrict__ Vtg)
{
  __shared__ __align__(16) unsigned short As[128 * 32];   // [row][k] packed
  __shared__ __align__(16) unsigned short BsT[64 * 32];   // [n][k] transposed
  const int t = threadIdx.x, lane = t & 31, wid = t >> 5;
  const int m0 = blockIdx.x * 128, n0 = blockIdx.y * 64;
  const int wm = (wid & 3) * 32, wn = (wid >> 2) * 32;

  FragF32 acc[2][2];
  #pragma unroll
  for (int i = 0; i < 2; i++)
    #pragma unroll
    for (int j = 0; j < 2; j++)
      #pragma unroll
      for (int r = 0; r < 8; r++) acc[i][j].f[r] = 0.f;

  const int am = lane & 15, ak = (lane >> 4) * 16;
  const int bn = lane & 15, bk = (lane >> 4) * 16;

  for (int kk = 0; kk < DD; kk += 32) {
    // Stage A (128x32): 1024 float4 chunks, 4 per thread, b128 loads.
    #pragma unroll
    for (int i = 0; i < 4; i++) {
      int idx = i * 256 + t, r = idx >> 3, c4 = idx & 7;
      float4 x = *(const float4*)&X[(size_t)(m0 + r) * DD + kk + c4 * 4];
      unsigned* dst = (unsigned*)&As[r * 32 + c4 * 4];
      dst[0] = packbf(x.x, x.y);
      dst[1] = packbf(x.z, x.w);
    }
    // Stage B (32x64) transposed into BsT[n][k]: 512 float4, 2 per thread.
    #pragma unroll
    for (int i = 0; i < 2; i++) {
      int idx = i * 256 + t, r = idx >> 4, c4 = idx & 15;
      float4 w = *(const float4*)&W[(size_t)(kk + r) * N1 + n0 + c4 * 4];
      unsigned p01 = packbf(w.x, w.y), p23 = packbf(w.z, w.w);
      BsT[(c4 * 4 + 0) * 32 + r] = (unsigned short)p01;
      BsT[(c4 * 4 + 1) * 32 + r] = (unsigned short)(p01 >> 16);
      BsT[(c4 * 4 + 2) * 32 + r] = (unsigned short)p23;
      BsT[(c4 * 4 + 3) * 32 + r] = (unsigned short)(p23 >> 16);
    }
    if (kk + 32 < DD) {      // global_prefetch_b8 of next K-step
      __builtin_prefetch(&X[(size_t)(m0 + (t >> 3)) * DD + kk + 32], 0, 1);
      __builtin_prefetch(&W[(size_t)(kk + 32 + (t >> 6)) * N1 + n0], 0, 1);
    }
    __syncthreads();

    FragB16 af[2], bfr[2];
    #pragma unroll
    for (int i = 0; i < 2; i++) {
      const unsigned short* ap = &As[(wm + i * 16 + am) * 32];
      load_frag(af[i], ap + ak, ap + ak + 8);
    }
    #pragma unroll
    for (int j = 0; j < 2; j++) {
      const unsigned short* bp = &BsT[(wn + j * 16 + bn) * 32];
      load_frag(bfr[j], bp + bk, bp + bk + 8);
    }
    #pragma unroll
    for (int i = 0; i < 2; i++)
      #pragma unroll
      for (int j = 0; j < 2; j++)
        acc[i][j].v = wmma_bf16(af[i], bfr[j], acc[i][j].v);
    __syncthreads();
  }

  const float scale = rsqrtf((float)HD) / ((float)(*layer_idx) + 1.f);
  #pragma unroll
  for (int i = 0; i < 2; i++) {
    #pragma unroll
    for (int j = 0; j < 2; j++) {
      int n = n0 + wn + j * 16 + (lane & 15);
      float bv = bias[n];
      int q3 = n >> 11;          // which of q/k/v (uniform per 16-col tile)
      int h  = (n >> 7) & 15;    // head
      int d  = n & 127;          // head dim
      #pragma unroll
      for (int r = 0; r < 8; r++) {
        int mrow = m0 + wm + i * 16 + (lane >> 4) * 8 + r;
        int b = mrow >> 11, s = mrow & 2047;
        float val = acc[i][j].f[r] + bv;
        size_t bh = (size_t)(b * HH + h);
        if (q3 == 0)      Qg[(bh * SS + s) * HD + d] = f2bf(val * scale);
        else if (q3 == 1) Kg[(bh * SS + s) * HD + d] = f2bf(val);
        else              Vtg[(bh * HD + d) * SS + s] = f2bf(val);
      }
    }
  }
}

// ---------------------------------------------------------------------------
// Kernel 2: causal flash attention. One wave owns 16 query rows.
// Transposed scores: S^T = K_tile(16k x 128d) * Q^T(128d x 16q), so each
// query's softmax stats are lane-local modulo one shfl_xor(16).
// Then O^T(16d x 16q) += V^T(16d x 32k) * P^T(32k x 16q) per d-chunk.
// All fragment loads are contiguous b128 global loads by layout design.
// ---------------------------------------------------------------------------
__global__ __launch_bounds__(256) void attn_kernel(
    const unsigned short* __restrict__ Qg, const unsigned short* __restrict__ Kg,
    const unsigned short* __restrict__ Vtg, unsigned short* __restrict__ Og)
{
  const int t = threadIdx.x, lane = t & 31, wid = t >> 5;
  const int h = blockIdx.y, b = blockIdx.z;
  const int qbase = (blockIdx.x * 8 + wid) * 16;
  const size_t bh = (size_t)(b * HH + h);
  const unsigned short* Qp = Qg + bh * SS * HD;
  const unsigned short* Kp = Kg + bh * SS * HD;
  const unsigned short* Vp = Vtg + bh * HD * SS;

  const int qcol = lane & 15;
  const int hi = lane >> 4;
  const int q = qbase + qcol;

  // Q fragments (B role), 4 chunks along d. Layout k=(hi*16+2v) -> contiguous.
  FragB16 qf[4];
  #pragma unroll
  for (int c = 0; c < 4; c++) {
    const unsigned short* p = Qp + (size_t)q * HD + c * 32 + hi * 16;
    load_frag(qf[c], p, p + 8);
  }

  FragF32 ot[8];
  #pragma unroll
  for (int dc = 0; dc < 8; dc++)
    #pragma unroll
    for (int r = 0; r < 8; r++) ot[dc].f[r] = 0.f;
  float mrun = -1e30f, lrun = 0.f;

  const int ktend = (qbase + 15) / 32 + 1;
  for (int kt = 0; kt < ktend; kt++) {
    const int kb = kt * 32;
    FragF32 sta, stb;                       // keys kb..+15 / kb+16..+31
    #pragma unroll
    for (int r = 0; r < 8; r++) { sta.f[r] = 0.f; stb.f[r] = 0.f; }
    #pragma unroll
    for (int c = 0; c < 4; c++) {
      FragB16 ka, kb2;
      const unsigned short* pa0 = Kp + (size_t)(kb + (lane & 15)) * HD + c * 32 + hi * 16;
      const unsigned short* pb0 = pa0 + (size_t)16 * HD;
      load_frag(ka,  pa0, pa0 + 8);
      load_frag(kb2, pb0, pb0 + 8);
      sta.v = wmma_bf16(ka, qf[c], sta.v);
      stb.v = wmma_bf16(kb2, qf[c], stb.v);
    }
    // Causal mask (selects, no branches) + online softmax update.
    float pa[8], pb[8], tmax = -1e30f;
    #pragma unroll
    for (int r = 0; r < 8; r++) {
      int kia = kb + hi * 8 + r, kib = kia + 16;
      pa[r] = (kia <= q) ? sta.f[r] : -1e30f;
      pb[r] = (kib <= q) ? stb.f[r] : -1e30f;
      tmax = fmaxf(tmax, fmaxf(pa[r], pb[r]));
    }
    tmax = fmaxf(tmax, __shfl_xor(tmax, 16, 32));
    float mnew = fmaxf(mrun, tmax);
    float alpha = __expf(mrun - mnew);
    float psum = 0.f;
    #pragma unroll
    for (int r = 0; r < 8; r++) {
      pa[r] = __expf(pa[r] - mnew);
      pb[r] = __expf(pb[r] - mnew);
      psum += pa[r] + pb[r];
    }
    psum += __shfl_xor(psum, 16, 32);
    lrun = lrun * alpha + psum;
    mrun = mnew;
    // Repack P^T into the 32x16 B-fragment layout (one half-swap shuffle).
    float opa[8], opb[8];
    #pragma unroll
    for (int r = 0; r < 8; r++) {
      opa[r] = __shfl_xor(pa[r], 16, 32);
      opb[r] = __shfl_xor(pb[r], 16, 32);
    }
    FragB16 pf;
    #pragma unroll
    for (int v = 0; v < 4; v++) {
      float l0 = (hi == 0) ? pa[2 * v]      : opb[2 * v];
      float l1 = (hi == 0) ? pa[2 * v + 1]  : opb[2 * v + 1];
      float h0 = (hi == 0) ? opa[2 * v]     : pb[2 * v];
      float h1 = (hi == 0) ? opa[2 * v + 1] : pb[2 * v + 1];
      pf.u[v]     = packbf(l0, l1);
      pf.u[v + 4] = packbf(h0, h1);
    }
    // Rescale accumulators, then O^T += V^T * P^T (8 d-chunks).
    #pragma unroll
    for (int dc = 0; dc < 8; dc++)
      #pragma unroll
      for (int r = 0; r < 8; r++) ot[dc].f[r] *= alpha;
    #pragma unroll
    for (int dc = 0; dc < 8; dc++) {
      FragB16 vf;
      const unsigned short* pv = Vp + (size_t)(dc * 16 + (lane & 15)) * SS + kb + hi * 16;
      load_frag(vf, pv, pv + 8);
      ot[dc].v = wmma_bf16(vf, pf, ot[dc].v);
    }
  }

  // Normalize and store O[b*S+q][h*128+d] as bf16 (8 halves = one b128 store).
  const float invl = 1.f / lrun;
  unsigned short* orow = Og + ((size_t)(b * SS + q) * (HH * HD)) + h * HD;
  #pragma unroll
  for (int dc = 0; dc < 8; dc++) {
    uint4v pk;
    pk.x = packbf(ot[dc].f[0] * invl, ot[dc].f[1] * invl);
    pk.y = packbf(ot[dc].f[2] * invl, ot[dc].f[3] * invl);
    pk.z = packbf(ot[dc].f[4] * invl, ot[dc].f[5] * invl);
    pk.w = packbf(ot[dc].f[6] * invl, ot[dc].f[7] * invl);
    *(uint4v*)(orow + dc * 16 + hi * 8) = pk;
  }
}

// ---------------------------------------------------------------------------
// Kernel 3: output projection. O[4096][2048] bf16 * Wp[2048][2048] + bias,
// f32 output. Same block-GEMM skeleton as kernel 1.
// ---------------------------------------------------------------------------
__global__ __launch_bounds__(256) void proj_kernel(
    const unsigned short* __restrict__ Og, const float* __restrict__ Wp,
    const float* __restrict__ bp, float* __restrict__ out)
{
  __shared__ __align__(16) unsigned short As[128 * 32];
  __shared__ __align__(16) unsigned short BsT[64 * 32];
  const int t = threadIdx.x, lane = t & 31, wid = t >> 5;
  const int m0 = blockIdx.x * 128, n0 = blockIdx.y * 64;
  const int wm = (wid & 3) * 32, wn = (wid >> 2) * 32;

  FragF32 acc[2][2];
  #pragma unroll
  for (int i = 0; i < 2; i++)
    #pragma unroll
    for (int j = 0; j < 2; j++)
      #pragma unroll
      for (int r = 0; r < 8; r++) acc[i][j].f[r] = 0.f;

  const int am = lane & 15, ak = (lane >> 4) * 16;
  const int bn = lane & 15, bk = (lane >> 4) * 16;

  for (int kk = 0; kk < DD; kk += 32) {
    // Stage A (128x32 bf16): straight b128 copies, 2 per thread.
    #pragma unroll
    for (int i = 0; i < 2; i++) {
      int idx = i * 256 + t, r = idx >> 2, c4 = idx & 3;
      *(uint4v*)&As[r * 32 + c4 * 8] =
          *(const uint4v*)&Og[(size_t)(m0 + r) * DD + kk + c4 * 8];
    }
    // Stage B (32x64) transposed into BsT[n][k].
    #pragma unroll
    for (int i = 0; i < 2; i++) {
      int idx = i * 256 + t, r = idx >> 4, c4 = idx & 15;
      float4 w = *(const float4*)&Wp[(size_t)(kk + r) * DD + n0 + c4 * 4];
      unsigned p01 = packbf(w.x, w.y), p23 = packbf(w.z, w.w);
      BsT[(c4 * 4 + 0) * 32 + r] = (unsigned short)p01;
      BsT[(c4 * 4 + 1) * 32 + r] = (unsigned short)(p01 >> 16);
      BsT[(c4 * 4 + 2) * 32 + r] = (unsigned short)p23;
      BsT[(c4 * 4 + 3) * 32 + r] = (unsigned short)(p23 >> 16);
    }
    if (kk + 32 < DD)
      __builtin_prefetch(&Wp[(size_t)(kk + 32 + (t >> 6)) * DD + n0], 0, 1);
    __syncthreads();

    FragB16 af[2], bfr[2];
    #pragma unroll
    for (int i = 0; i < 2; i++) {
      const unsigned short* ap = &As[(wm + i * 16 + am) * 32];
      load_frag(af[i], ap + ak, ap + ak + 8);
    }
    #pragma unroll
    for (int j = 0; j < 2; j++) {
      const unsigned short* bp2 = &BsT[(wn + j * 16 + bn) * 32];
      load_frag(bfr[j], bp2 + bk, bp2 + bk + 8);
    }
    #pragma unroll
    for (int i = 0; i < 2; i++)
      #pragma unroll
      for (int j = 0; j < 2; j++)
        acc[i][j].v = wmma_bf16(af[i], bfr[j], acc[i][j].v);
    __syncthreads();
  }

  #pragma unroll
  for (int i = 0; i < 2; i++) {
    #pragma unroll
    for (int j = 0; j < 2; j++) {
      int n = n0 + wn + j * 16 + (lane & 15);
      float bv = bp[n];
      #pragma unroll
      for (int r = 0; r < 8; r++) {
        int mrow = m0 + wm + i * 16 + (lane >> 4) * 8 + r;
        out[(size_t)mrow * DD + n] = acc[i][j].f[r] + bv;
      }
    }
  }
}

// ---------------------------------------------------------------------------
extern "C" void kernel_launch(void* const* d_in, const int* in_sizes, int n_in,
                              void* d_out, int out_size, void* d_ws, size_t ws_size,
                              hipStream_t stream) {
  const float* X  = (const float*)d_in[0];   // hidden_states [B,S,D]
  const float* W  = (const float*)d_in[1];   // w_attn [D,3,H,HD]
  const float* ba = (const float*)d_in[2];   // b_attn [3,H,HD]
  const float* Wp = (const float*)d_in[3];   // w_proj [H,HD,D]
  const float* bp = (const float*)d_in[4];   // b_proj [D]
  const int*   li = (const int*)d_in[5];     // layer_idx

  char* ws = (char*)d_ws;
  const size_t seg = (size_t)BB * HH * SS * HD * sizeof(unsigned short); // 16.78MB
  unsigned short* Qg  = (unsigned short*)(ws);
  unsigned short* Kg  = (unsigned short*)(ws + seg);
  unsigned short* Vtg = (unsigned short*)(ws + 2 * seg);
  unsigned short* Og  = (unsigned short*)(ws + 3 * seg);

  qkv_kernel <<<dim3(M1 / 128, N1 / 64), 256, 0, stream>>>(X, W, ba, li, Qg, Kg, Vtg);
  attn_kernel<<<dim3(SS / 128, HH, BB),  256, 0, stream>>>(Qg, Kg, Vtg, Og);
  proj_kernel<<<dim3(M1 / 128, DD / 64), 256, 0, stream>>>(Og, Wp, bp, (float*)d_out);
}
